// MultiHeadAttention_33105607917987
// MI455X (gfx1250) — matrix operand order repaired and checked by reading, not compiled, not run on hardware
//
#include <hip/hip_runtime.h>
#include <hip/hip_bf16.h>

typedef _Float16 v4h  __attribute__((ext_vector_type(4)));
typedef _Float16 v8h  __attribute__((ext_vector_type(8)));
typedef _Float16 v16h __attribute__((ext_vector_type(16)));
typedef float    v8f  __attribute__((ext_vector_type(8)));
typedef unsigned int u32x4 __attribute__((ext_vector_type(4)));
typedef int          i32x4 __attribute__((ext_vector_type(4)));
typedef int          i32x8 __attribute__((ext_vector_type(8)));

#define B_   2
#define S_   2048
#define D_   1024
#define H_   16
#define DH_  64
#define M_   (B_ * S_)      // 4096 tokens
#define N3_  (3 * D_)       // 3072

static __device__ __forceinline__ v16h frag_combine(v8h lo, v8h hi) {
    v16h r;
#pragma unroll
    for (int i = 0; i < 8; ++i) { r[i] = lo[i]; r[i + 8] = hi[i]; }
    return r;
}

// -------- Tensor Data Mover: load a rows x 32 f16 tile (row stride = stride
// elements) from global into LDS at lds_addr, laid out row-major [rows][32].
// D# bit layout per CDNA5 ISA (group0 + group1; groups
// 2/3 zero => dims 2..4 unused). 6-arg builtin form (clang-23 toolchain).
static __device__ __forceinline__ void tdm_load_tile_f16(
        unsigned lds_addr, const void* gptr, int rows, int stride_elems) {
    unsigned long long ga = (unsigned long long)(size_t)gptr;
    u32x4 g0;
    g0[0] = 1u;                                   // count=1, is_restore=0, no gather
    g0[1] = lds_addr;                             // bits 63:32  lds_addr
    g0[2] = (unsigned)ga;                         // bits 95:64  global_addr[31:0]
    g0[3] = (unsigned)(ga >> 32) | (2u << 30);    // global_addr[56:32] | type=2

    const unsigned td0 = 1u << 20;                // oversized tensor dims: no OOB clip
    const unsigned td1 = 1u << 20;
    i32x8 g1;
    g1[0] = 1 << 16;                              // data_size=1 (2 bytes)
    g1[1] = (int)((td0 & 0xFFFFu) << 16);         // tensor_dim0[15:0]
    g1[2] = (int)((td0 >> 16) | ((td1 & 0xFFFFu) << 16));
    g1[3] = (int)((td1 >> 16) | (32u << 16));     // tensor_dim1[31:16] | tile_dim0=32
    g1[4] = rows;                                 // tile_dim1 | tile_dim2=0
    g1[5] = stride_elems;                         // tensor_dim0_stride[31:0]
    g1[6] = 0;                                    // stride[47:32] | dim1_stride[15:0]
    g1[7] = 0;

    i32x4 z4 = {0, 0, 0, 0};
    i32x8 z8 = {0, 0, 0, 0, 0, 0, 0, 0};
    __builtin_amdgcn_tensor_load_to_lds(g0, g1, z4, z4, z8, 0);
}

// ---------------------------------------------------------------- fp32 -> f16
__global__ __launch_bounds__(256) void mha_cvt_f16(const float* __restrict__ s,
                                                   _Float16* __restrict__ d, int n) {
    int i = 4 * (blockIdx.x * 256 + threadIdx.x);
    if (i >= n) return;
    float4 v = *(const float4*)(s + i);
    v4h o = { (_Float16)v.x, (_Float16)v.y, (_Float16)v.z, (_Float16)v.w };
    *(v4h*)(d + i) = o;
}

// ------------------------------------------------- QKV projection GEMM (WMMA)
// C[M=4096, N=3072] = Xh @ Wqkvh^T + bqkv ; scatter f16 into Q/K/V [B,H,S,Dh]
__global__ __launch_bounds__(256) void mha_qkv_gemm(
        const _Float16* __restrict__ X, const _Float16* __restrict__ W,
        const float* __restrict__ bias,
        _Float16* __restrict__ Qo, _Float16* __restrict__ Ko, _Float16* __restrict__ Vo) {
    __shared__ __align__(16) _Float16 As[128 * 32];
    __shared__ __align__(16) _Float16 Bs[64 * 32];

    const int t = threadIdx.x, wave = t >> 5, lane = t & 31;
    const int lh = lane >> 4, ln = lane & 15;
    const int wm = wave >> 1, wn = wave & 1;           // 4x2 wave grid
    const int m0 = blockIdx.y * 128, n0 = blockIdx.x * 64;
    const unsigned ldsA = (unsigned)(size_t)&As[0];
    const unsigned ldsB = (unsigned)(size_t)&Bs[0];

    v8f acc[2][2] = {};
    for (int k0 = 0; k0 < D_; k0 += 32) {
        __syncthreads();                               // LDS free from last iter
        if (wave == 0) {                               // TDM DMA both tiles
            tdm_load_tile_f16(ldsA, &X[(size_t)m0 * D_ + k0], 128, D_);
            tdm_load_tile_f16(ldsB, &W[(size_t)n0 * D_ + k0], 64, D_);
            __builtin_amdgcn_s_wait_tensorcnt(0);
        }
        __syncthreads();                               // tiles visible to all waves

        v16h af[2], bf[2];
#pragma unroll
        for (int s = 0; s < 2; ++s) {
            const _Float16* pa = &As[(wm * 32 + s * 16 + ln) * 32 + 8 * lh];
            af[s] = frag_combine(*(const v8h*)pa, *(const v8h*)(pa + 16));
            const _Float16* pb = &Bs[(wn * 32 + s * 16 + ln) * 32 + 8 * lh];
            bf[s] = frag_combine(*(const v8h*)pb, *(const v8h*)(pb + 16));
        }
#pragma unroll
        for (int si = 0; si < 2; ++si)
#pragma unroll
            for (int sj = 0; sj < 2; ++sj)
                acc[si][sj] = __builtin_amdgcn_wmma_f32_16x16x32_f16(
                    false, af[si], false, bf[sj], (short)0, acc[si][sj], false, false);
    }

    // epilogue: +bias, scatter to head-major f16 Q/K/V
#pragma unroll
    for (int si = 0; si < 2; ++si)
#pragma unroll
        for (int sj = 0; sj < 2; ++sj) {
            int mb = m0 + wm * 32 + si * 16;
            int n  = n0 + wn * 32 + sj * 16 + ln;
            float bn = bias[n];
            int which = n >> 10, dcol = n & 1023;
            int h = dcol >> 6, dh = dcol & 63;
            _Float16* dst = (which == 0) ? Qo : ((which == 1) ? Ko : Vo);
#pragma unroll
            for (int r = 0; r < 8; ++r) {
                int m = mb + r + 8 * lh;
                int bb = m >> 11, ss = m & (S_ - 1);
                dst[(((size_t)bb * H_ + h) * S_ + ss) * DH_ + dh] =
                    (_Float16)(acc[si][sj][r] + bn);
            }
        }
}

// ------------------------------------------------------ causal flash attention
// grid (S/64, H, B), block 128 = 4 waves; wave w owns query rows q0..q0+15
__global__ __launch_bounds__(128) void mha_attn(
        const _Float16* __restrict__ Qh, const _Float16* __restrict__ Kh,
        const _Float16* __restrict__ Vh, _Float16* __restrict__ Ao) {
    __shared__ __align__(16) _Float16 Ks[32 * 64];     // [key][dh]
    __shared__ __align__(16) _Float16 VTs[64 * 32];    // [dh][key]
    __shared__ __align__(16) _Float16 Ps[4][16 * 32];  // per-wave P staging

    const int t = threadIdx.x, wave = t >> 5, lane = t & 31;
    const int lh = lane >> 4, ln = lane & 15;
    const int qblk = blockIdx.x, h = blockIdx.y, b = blockIdx.z;
    const int q0 = qblk * 64 + wave * 16;
    const size_t base = ((size_t)b * H_ + h) * S_ * DH_;
    const float scale = 0.125f;                        // 1/sqrt(64)

    v16h qf[2];
    {
        const _Float16* qrow = Qh + base + (size_t)(q0 + ln) * DH_;
#pragma unroll
        for (int d = 0; d < 2; ++d) {
            const _Float16* p = qrow + d * 32 + 8 * lh;
            qf[d] = frag_combine(*(const v8h*)p, *(const v8h*)(p + 16));
        }
    }

    v8f o[4] = {};
    float mrow[8], lrow[8];
#pragma unroll
    for (int r = 0; r < 8; ++r) { mrow[r] = -3.0e38f; lrow[r] = 0.0f; }

    const int ntiles = (qblk + 1) * 2;                 // 32-key tiles up to causal edge
    for (int kt = 0; kt < ntiles; ++kt) {
        const int kb = kt * 32;
        __syncthreads();
#pragma unroll
        for (int i = 0; i < 2; ++i) {                  // K tile 32x64
            int idx = t + i * 128, row = idx >> 3, c8 = (idx & 7) * 8;
            *(v8h*)&Ks[row * 64 + c8] =
                *(const v8h*)&Kh[base + (size_t)(kb + row) * DH_ + c8];
        }
#pragma unroll
        for (int i = 0; i < 2; ++i) {                  // V tile 32x64 transposed into LDS
            int idx = t + i * 128, row = idx >> 3, c8 = (idx & 7) * 8;
            v8h v = *(const v8h*)&Vh[base + (size_t)(kb + row) * DH_ + c8];
#pragma unroll
            for (int j = 0; j < 8; ++j) VTs[(c8 + j) * 32 + row] = v[j];
        }
        __syncthreads();

        v8f sc[2] = {};                                // scores: 16 rows x 32 keys
#pragma unroll
        for (int c = 0; c < 2; ++c)
#pragma unroll
            for (int d = 0; d < 2; ++d) {
                const _Float16* p = &Ks[(16 * c + ln) * 64 + 32 * d + 8 * lh];
                v16h bf = frag_combine(*(const v8h*)p, *(const v8h*)(p + 16));
                sc[c] = __builtin_amdgcn_wmma_f32_16x16x32_f16(
                    false, qf[d], false, bf, (short)0, sc[c], false, false);
            }

#pragma unroll
        for (int c = 0; c < 2; ++c) {                  // scale + causal mask (-1e9)
            int key = kb + 16 * c + ln;
#pragma unroll
            for (int r = 0; r < 8; ++r) {
                int row = q0 + r + 8 * lh;
                float v = sc[c][r] * scale;
                sc[c][r] = (key <= row) ? v : -1.0e9f;
            }
        }

        float pold[8];
#pragma unroll
        for (int r = 0; r < 8; ++r) {                  // row max over 16 lanes of half
            float v = fmaxf(sc[0][r], sc[1][r]);
#pragma unroll
            for (int msk = 1; msk < 16; msk <<= 1) v = fmaxf(v, __shfl_xor(v, msk, 32));
            float mnew = fmaxf(mrow[r], v);
            pold[r] = __expf(mrow[r] - mnew);
            mrow[r] = mnew;
        }
#pragma unroll
        for (int c = 0; c < 2; ++c)
#pragma unroll
            for (int r = 0; r < 8; ++r) sc[c][r] = __expf(sc[c][r] - mrow[r]);
#pragma unroll
        for (int r = 0; r < 8; ++r) {                  // row sum + rescale state
            float v = sc[0][r] + sc[1][r];
#pragma unroll
            for (int msk = 1; msk < 16; msk <<= 1) v += __shfl_xor(v, msk, 32);
            lrow[r] = lrow[r] * pold[r] + v;
        }
#pragma unroll
        for (int nc = 0; nc < 4; ++nc)
#pragma unroll
            for (int r = 0; r < 8; ++r) o[nc][r] *= pold[r];

        _Float16* P = &Ps[wave][0];                    // C-layout -> A-layout via LDS
#pragma unroll
        for (int c = 0; c < 2; ++c)
#pragma unroll
            for (int r = 0; r < 8; ++r)
                P[(r + 8 * lh) * 32 + 16 * c + ln] = (_Float16)sc[c][r];

        const _Float16* pp = &P[ln * 32 + 8 * lh];
        v16h pf = frag_combine(*(const v8h*)pp, *(const v8h*)(pp + 16));
#pragma unroll
        for (int nc = 0; nc < 4; ++nc) {
            const _Float16* pv = &VTs[(16 * nc + ln) * 32 + 8 * lh];
            v16h vf = frag_combine(*(const v8h*)pv, *(const v8h*)(pv + 16));
            o[nc] = __builtin_amdgcn_wmma_f32_16x16x32_f16(
                false, pf, false, vf, (short)0, o[nc], false, false);
        }
    }

    // normalize and store f16 attn output [B*S, D] row-major
#pragma unroll
    for (int nc = 0; nc < 4; ++nc) {
        int dh = 16 * nc + ln;
#pragma unroll
        for (int r = 0; r < 8; ++r) {
            int row = q0 + r + 8 * lh;
            Ao[((size_t)b * S_ + row) * D_ + h * DH_ + dh] =
                (_Float16)(o[nc][r] / lrow[r]);
        }
    }
}

// ---------------------------------------------- output projection GEMM (WMMA)
// out[M=4096, N=1024] = Ah @ Woh^T + bo  (fp32 store)
__global__ __launch_bounds__(256) void mha_out_gemm(
        const _Float16* __restrict__ X, const _Float16* __restrict__ W,
        const float* __restrict__ bias, float* __restrict__ out) {
    __shared__ __align__(16) _Float16 As[128 * 32];
    __shared__ __align__(16) _Float16 Bs[64 * 32];

    const int t = threadIdx.x, wave = t >> 5, lane = t & 31;
    const int lh = lane >> 4, ln = lane & 15;
    const int wm = wave >> 1, wn = wave & 1;
    const int m0 = blockIdx.y * 128, n0 = blockIdx.x * 64;
    const unsigned ldsA = (unsigned)(size_t)&As[0];
    const unsigned ldsB = (unsigned)(size_t)&Bs[0];

    v8f acc[2][2] = {};
    for (int k0 = 0; k0 < D_; k0 += 32) {
        __syncthreads();
        if (wave == 0) {
            tdm_load_tile_f16(ldsA, &X[(size_t)m0 * D_ + k0], 128, D_);
            tdm_load_tile_f16(ldsB, &W[(size_t)n0 * D_ + k0], 64, D_);
            __builtin_amdgcn_s_wait_tensorcnt(0);
        }
        __syncthreads();

        v16h af[2], bf[2];
#pragma unroll
        for (int s = 0; s < 2; ++s) {
            const _Float16* pa = &As[(wm * 32 + s * 16 + ln) * 32 + 8 * lh];
            af[s] = frag_combine(*(const v8h*)pa, *(const v8h*)(pa + 16));
            const _Float16* pb = &Bs[(wn * 32 + s * 16 + ln) * 32 + 8 * lh];
            bf[s] = frag_combine(*(const v8h*)pb, *(const v8h*)(pb + 16));
        }
#pragma unroll
        for (int si = 0; si < 2; ++si)
#pragma unroll
            for (int sj = 0; sj < 2; ++sj)
                acc[si][sj] = __builtin_amdgcn_wmma_f32_16x16x32_f16(
                    false, af[si], false, bf[sj], (short)0, acc[si][sj], false, false);
    }

#pragma unroll
    for (int si = 0; si < 2; ++si)
#pragma unroll
        for (int sj = 0; sj < 2; ++sj) {
            int mb = m0 + wm * 32 + si * 16;
            int n  = n0 + wn * 32 + sj * 16 + ln;
            float bn = bias[n];
#pragma unroll
            for (int r = 0; r < 8; ++r) {
                int m = mb + r + 8 * lh;
                out[(size_t)m * D_ + n] = acc[si][sj][r] + bn;
            }
        }
}

// ---------------------------------------------------------------------- launch
extern "C" void kernel_launch(void* const* d_in, const int* in_sizes, int n_in,
                              void* d_out, int out_size, void* d_ws, size_t ws_size,
                              hipStream_t stream) {
    const float* hs   = (const float*)d_in[0];   // [2,2048,1024]
    const float* Wqkv = (const float*)d_in[1];   // [3072,1024]
    const float* bqkv = (const float*)d_in[2];   // [3072]
    const float* Wo   = (const float*)d_in[3];   // [1024,1024]
    const float* bo   = (const float*)d_in[4];   // [1024]
    float* out = (float*)d_out;

    char* ws = (char*)d_ws;
    _Float16* Xh    = (_Float16*)(ws);                    //  8 MiB
    _Float16* Wqkvh = (_Float16*)(ws + 8388608);          //  6 MiB
    _Float16* Woh   = (_Float16*)(ws + 14680064);         //  2 MiB
    _Float16* Qh    = (_Float16*)(ws + 16777216);         //  8 MiB
    _Float16* Kh    = (_Float16*)(ws + 25165824);         //  8 MiB
    _Float16* Vh    = (_Float16*)(ws + 33554432);         //  8 MiB
    _Float16* Ah    = (_Float16*)(ws + 41943040);         //  8 MiB

    mha_cvt_f16<<<(M_ * D_) / 1024, 256, 0, stream>>>(hs, Xh, M_ * D_);
    mha_cvt_f16<<<(N3_ * D_) / 1024, 256, 0, stream>>>(Wqkv, Wqkvh, N3_ * D_);
    mha_cvt_f16<<<(D_ * D_) / 1024, 256, 0, stream>>>(Wo, Woh, D_ * D_);

    mha_qkv_gemm<<<dim3(N3_ / 64, M_ / 128), 256, 0, stream>>>(
        Xh, Wqkvh, bqkv, Qh, Kh, Vh);

    mha_attn<<<dim3(S_ / 64, H_, B_), 128, 0, stream>>>(Qh, Kh, Vh, Ah);

    mha_out_gemm<<<dim3(D_ / 64, M_ / 128), 256, 0, stream>>>(Ah, Woh, bo, out);
}
